// MultiHeadAttention_7026566496597
// MI455X (gfx1250) — compile-verified, matrix operand order checked by reading, not run on hardware
//
#include <hip/hip_runtime.h>
#include <hip/hip_bf16.h>

#define SEQ_LEN 2048
#define BATCH_N 4
#define D_MODEL 1024
#define N_HEADS 16
#define D_K     64
#define SCALE_F 0.125f   // 1/sqrt(64)

#define KSTR 72          // K-tile LDS row stride in halves (144B: 16B aligned, conflict-free)
#define VSTR 40          // V-tile LDS row stride in halves (80B:  16B aligned, conflict-free)

typedef _Float16 h4   __attribute__((ext_vector_type(4)));
typedef _Float16 h8   __attribute__((ext_vector_type(8)));
typedef _Float16 v16h __attribute__((ext_vector_type(16)));
typedef float    v8f  __attribute__((ext_vector_type(8)));

static __device__ __forceinline__ v16h mk16(h8 lo, h8 hi) {
  union { h8 p[2]; v16h v; } u;
  u.p[0] = lo; u.p[1] = hi;
  return u.v;
}

static __device__ __forceinline__ h8 ld_cvt8(const float* __restrict__ p) {
  h8 r;
#pragma unroll
  for (int i = 0; i < 8; ++i) r[i] = (_Float16)p[i];
  return r;
}

// Async global->LDS 16B copy (CDNA5, tracked by ASYNCcnt). Per-lane addresses.
static __device__ __forceinline__ void async_b128(const void* gaddr, void* lds) {
  unsigned ldsoff = (unsigned)(unsigned long long)lds;  // aperture low bits = LDS offset
  asm volatile("global_load_async_to_lds_b128 %0, %1, off"
               :: "v"(ldsoff), "v"(gaddr) : "memory");
}
static __device__ __forceinline__ void wait_async0() {
  asm volatile("s_wait_asynccnt 0" ::: "memory");
}

// ---------------------------------------------------------------------------
// f32 -> f16 weight conversion (weights reused 512x per tile-row, convert once)
// ---------------------------------------------------------------------------
__global__ void __launch_bounds__(256) wcvt_f16(const float* __restrict__ src,
                                                _Float16* __restrict__ dst, int n4) {
  int i = blockIdx.x * blockDim.x + threadIdx.x;
  if (i < n4) {
    const float4 v = ((const float4*)src)[i];
    h4 o;
    o[0] = (_Float16)v.x; o[1] = (_Float16)v.y;
    o[2] = (_Float16)v.z; o[3] = (_Float16)v.w;
    ((h4*)dst)[i] = o;
  }
}

// ---------------------------------------------------------------------------
// GEMM: C[M,N] = A[M,K](f32, converted in-register) * W[N,K]^T (f16) + bias
// One 16x64 C strip per wave (A-frag reused across 4 wmmas), 4 waves/block.
// mode 0: store f16 as Q/K layout [b*16+h][s][dk]
// mode 1: store f16 as V  layout [b*16+h][dk][s]   (transposed for PV B-frags)
// mode 2: store f32 row-major [m][n] (final output)
// ---------------------------------------------------------------------------
__global__ void __launch_bounds__(128) gemm_wmma(const float* __restrict__ A,
                                                 const _Float16* __restrict__ W,
                                                 const float* __restrict__ bias,
                                                 void* __restrict__ out,
                                                 int M, int N, int K, int mode) {
  const int lane = threadIdx.x & 31;
  const int wave = threadIdx.x >> 5;
  const int l15  = lane & 15;
  const int hi16 = lane >> 4;
  const int kb   = hi16 * 8;

  const int mtile = blockIdx.x * 16;
  const int ntile = blockIdx.y * 256 + wave * 64;

  const float*    arow = A + (size_t)(mtile + l15) * K;        // A-frag: lane = row
  const _Float16* b0   = W + (size_t)(ntile +  0 + l15) * K;   // B-frag: lane = col n
  const _Float16* b1   = W + (size_t)(ntile + 16 + l15) * K;
  const _Float16* b2   = W + (size_t)(ntile + 32 + l15) * K;
  const _Float16* b3   = W + (size_t)(ntile + 48 + l15) * K;

  v8f c0 = {}, c1 = {}, c2 = {}, c3 = {};
  for (int k0 = 0; k0 < K; k0 += 128) {
    __builtin_prefetch(arow + k0 + 512, 0, 1);
    __builtin_prefetch(b0 + k0 + 512, 0, 1);
    __builtin_prefetch(b1 + k0 + 512, 0, 1);
    __builtin_prefetch(b2 + k0 + 512, 0, 1);
    __builtin_prefetch(b3 + k0 + 512, 0, 1);
#pragma unroll
    for (int kk = 0; kk < 128; kk += 32) {
      const int k = k0 + kk;
      const v16h a = mk16(ld_cvt8(arow + k + kb), ld_cvt8(arow + k + kb + 16));
      c0 = __builtin_amdgcn_wmma_f32_16x16x32_f16(false, a, false,
             mk16(*(const h8*)(b0 + k + kb), *(const h8*)(b0 + k + kb + 16)),
             (short)0, c0, false, false);
      c1 = __builtin_amdgcn_wmma_f32_16x16x32_f16(false, a, false,
             mk16(*(const h8*)(b1 + k + kb), *(const h8*)(b1 + k + kb + 16)),
             (short)0, c1, false, false);
      c2 = __builtin_amdgcn_wmma_f32_16x16x32_f16(false, a, false,
             mk16(*(const h8*)(b2 + k + kb), *(const h8*)(b2 + k + kb + 16)),
             (short)0, c2, false, false);
      c3 = __builtin_amdgcn_wmma_f32_16x16x32_f16(false, a, false,
             mk16(*(const h8*)(b3 + k + kb), *(const h8*)(b3 + k + kb + 16)),
             (short)0, c3, false, false);
    }
  }

  v8f cc[4] = { c0, c1, c2, c3 };
#pragma unroll
  for (int t = 0; t < 4; ++t) {
    const int   ng = ntile + t * 16 + l15;
    const float bb = bias[ng];
    if (mode == 2) {
      float* o = (float*)out;
#pragma unroll
      for (int r = 0; r < 8; ++r) {
        const int mg = mtile + r + hi16 * 8;
        o[(size_t)mg * N + ng] = cc[t][r] + bb;
      }
    } else {
      _Float16* o = (_Float16*)out;
      const int h = ng >> 6, dk = ng & 63;
#pragma unroll
      for (int r = 0; r < 8; ++r) {
        const int mg = mtile + r + hi16 * 8;
        const int s = mg >> 2, b = mg & 3;     // input rows are (s, b) pairs
        const float v = cc[t][r] + bb;
        const size_t idx = (mode == 0)
          ? (((size_t)(b * N_HEADS + h) * SEQ_LEN + s) * D_K + dk)
          : (((size_t)(b * N_HEADS + h) * D_K + dk) * SEQ_LEN + s);
        o[idx] = (_Float16)v;
      }
    }
  }
}

// ---------------------------------------------------------------------------
// Causal flash attention. grid = (SEQ/64, B*H), block = 128 (4 waves).
// K/V 32-key tiles are staged into LDS with double-buffered
// global_load_async_to_lds_b128 (ASYNCcnt) and shared by all 4 waves.
// Q/K layout [bh][s][dk]; V transposed [bh][dk][s]. P transposed C->A layout
// through a per-wave LDS tile (96B row stride: 16B-aligned, conflict-free).
// ---------------------------------------------------------------------------
__global__ void __launch_bounds__(128) attn_fwd(const _Float16* __restrict__ Qh,
                                                const _Float16* __restrict__ Kh,
                                                const _Float16* __restrict__ Vt,
                                                float* __restrict__ Xo) {
  __shared__ alignas(16) _Float16 kbuf[2][32][KSTR];
  __shared__ alignas(16) _Float16 vbuf[2][64][VSTR];
  __shared__ alignas(16) _Float16 pbuf[4][16][48];

  const int tid   = threadIdx.x;
  const int lane  = tid & 31;
  const int wave  = tid >> 5;
  const int l15   = lane & 15;
  const int hi16  = lane >> 4;
  const int kb    = hi16 * 8;
  const int bh    = blockIdx.y;
  const int qblk  = blockIdx.x * 64;
  const int qbase = qblk + wave * 16;

  // Q fragments (16 rows x 64 dk = two K=32 A-frags), held in registers.
  const _Float16* qrow = Qh + ((size_t)bh * SEQ_LEN + qbase + l15) * D_K;
  const v16h aq0 = mk16(*(const h8*)(qrow + kb),      *(const h8*)(qrow + kb + 16));
  const v16h aq1 = mk16(*(const h8*)(qrow + 32 + kb), *(const h8*)(qrow + 48 + kb));

  float mrow[8], lrow[8];
  v8f acc0 = {}, acc1 = {}, acc2 = {}, acc3 = {};
#pragma unroll
  for (int r = 0; r < 8; ++r) { mrow[r] = -3.0e38f; lrow[r] = 0.0f; }

  // cooperative async staging of one 32-key K tile + V tile (4 x b128 / thread)
  auto stage = [&](int j0, int buf) {
#pragma unroll
    for (int c = 0; c < 2; ++c) {                 // K: 32 rows x 8 chunks
      const int chunk = tid * 2 + c;
      const int row = chunk >> 3, col = (chunk & 7) * 8;
      async_b128(Kh + ((size_t)bh * SEQ_LEN + j0 + row) * D_K + col,
                 &kbuf[buf][row][col]);
    }
#pragma unroll
    for (int c = 0; c < 2; ++c) {                 // V: 64 rows x 4 chunks
      const int chunk = tid * 2 + c;
      const int row = chunk >> 2, col = (chunk & 3) * 8;
      async_b128(Vt + ((size_t)bh * D_K + row) * SEQ_LEN + j0 + col,
                 &vbuf[buf][row][col]);
    }
  };

  const int jend = qblk + 63;          // last key index any wave in block needs
  stage(0, 0);
  wait_async0();
  __syncthreads();

  int cur = 0;
  for (int j0 = 0; j0 <= jend; j0 += 32) {
    if (j0 + 32 <= jend) stage(j0 + 32, cur ^ 1);

    // ---- S = Q * K^T over 32 LDS-resident keys (two 16-wide sub-tiles) ----
    const _Float16* kr0 = &kbuf[cur][l15][0];
    const _Float16* kr1 = &kbuf[cur][l15 + 16][0];
    v8f s0 = {}, s1 = {};
    s0 = __builtin_amdgcn_wmma_f32_16x16x32_f16(false, aq0, false,
           mk16(*(const h8*)(kr0 + kb), *(const h8*)(kr0 + kb + 16)),
           (short)0, s0, false, false);
    s0 = __builtin_amdgcn_wmma_f32_16x16x32_f16(false, aq1, false,
           mk16(*(const h8*)(kr0 + 32 + kb), *(const h8*)(kr0 + 48 + kb)),
           (short)0, s0, false, false);
    s1 = __builtin_amdgcn_wmma_f32_16x16x32_f16(false, aq0, false,
           mk16(*(const h8*)(kr1 + kb), *(const h8*)(kr1 + kb + 16)),
           (short)0, s1, false, false);
    s1 = __builtin_amdgcn_wmma_f32_16x16x32_f16(false, aq1, false,
           mk16(*(const h8*)(kr1 + 32 + kb), *(const h8*)(kr1 + 48 + kb)),
           (short)0, s1, false, false);

    // ---- scale + causal mask (C layout: row = r + 8*hi16, col = l15) ----
#pragma unroll
    for (int r = 0; r < 8; ++r) {
      const int irow = qbase + r + hi16 * 8;
      s0[r] = (j0 + l15      <= irow) ? s0[r] * SCALE_F : -3.0e38f;
      s1[r] = (j0 + 16 + l15 <= irow) ? s1[r] * SCALE_F : -3.0e38f;
    }

    // ---- online softmax: per-row max/sum via 16-lane xor reductions ----
#pragma unroll
    for (int r = 0; r < 8; ++r) {
      float v = fmaxf(s0[r], s1[r]);
      v = fmaxf(v, __shfl_xor(v, 1, 32));
      v = fmaxf(v, __shfl_xor(v, 2, 32));
      v = fmaxf(v, __shfl_xor(v, 4, 32));
      v = fmaxf(v, __shfl_xor(v, 8, 32));
      const float mnew = fmaxf(mrow[r], v);
      const float f    = __expf(mrow[r] - mnew);
      mrow[r] = mnew;
      const float p0 = __expf(s0[r] - mnew);
      const float p1 = __expf(s1[r] - mnew);
      float rs = p0 + p1;
      rs += __shfl_xor(rs, 1, 32);
      rs += __shfl_xor(rs, 2, 32);
      rs += __shfl_xor(rs, 4, 32);
      rs += __shfl_xor(rs, 8, 32);
      lrow[r] = lrow[r] * f + rs;
      acc0[r] *= f; acc1[r] *= f; acc2[r] *= f; acc3[r] *= f;
      const int prow = r + hi16 * 8;
      pbuf[wave][prow][l15]      = (_Float16)p0;   // C layout -> LDS
      pbuf[wave][prow][l15 + 16] = (_Float16)p1;
    }

    // ---- reload P as an A-frag (16x32, K = key index) ----
    const v16h pf = mk16(*(const h8*)&pbuf[wave][l15][kb],
                         *(const h8*)&pbuf[wave][l15][kb + 16]);

    // ---- O += P * V  (V tile in LDS: rows = dk, contiguous keys) ----
    const _Float16* vr0 = &vbuf[cur][l15][0];
    const _Float16* vr1 = &vbuf[cur][16 + l15][0];
    const _Float16* vr2 = &vbuf[cur][32 + l15][0];
    const _Float16* vr3 = &vbuf[cur][48 + l15][0];
    acc0 = __builtin_amdgcn_wmma_f32_16x16x32_f16(false, pf, false,
             mk16(*(const h8*)(vr0 + kb), *(const h8*)(vr0 + kb + 16)),
             (short)0, acc0, false, false);
    acc1 = __builtin_amdgcn_wmma_f32_16x16x32_f16(false, pf, false,
             mk16(*(const h8*)(vr1 + kb), *(const h8*)(vr1 + kb + 16)),
             (short)0, acc1, false, false);
    acc2 = __builtin_amdgcn_wmma_f32_16x16x32_f16(false, pf, false,
             mk16(*(const h8*)(vr2 + kb), *(const h8*)(vr2 + kb + 16)),
             (short)0, acc2, false, false);
    acc3 = __builtin_amdgcn_wmma_f32_16x16x32_f16(false, pf, false,
             mk16(*(const h8*)(vr3 + kb), *(const h8*)(vr3 + kb + 16)),
             (short)0, acc3, false, false);

    wait_async0();          // next tile landed in LDS
    __syncthreads();        // all waves done with cur tile; next tile visible
    cur ^= 1;
  }

  // ---- normalize and store X in [s*B+b][h*64+dk] f32 for the final GEMM ----
  const int b_ = bh >> 4;
  const int h_ = bh & 15;
  float inv[8];
#pragma unroll
  for (int r = 0; r < 8; ++r) inv[r] = 1.0f / lrow[r];
#pragma unroll
  for (int r = 0; r < 8; ++r) {
    const int s = qbase + r + hi16 * 8;
    float* orow = Xo + (size_t)(s * BATCH_N + b_) * D_MODEL + h_ * D_K + l15;
    orow[0]  = acc0[r] * inv[r];
    orow[16] = acc1[r] * inv[r];
    orow[32] = acc2[r] * inv[r];
    orow[48] = acc3[r] * inv[r];
  }
}

// ---------------------------------------------------------------------------
extern "C" void kernel_launch(void* const* d_in, const int* in_sizes, int n_in,
                              void* d_out, int out_size, void* d_ws, size_t ws_size,
                              hipStream_t stream) {
  (void)in_sizes; (void)n_in; (void)out_size; (void)ws_size;
  const float* query = (const float*)d_in[0];
  const float* key_  = (const float*)d_in[1];
  const float* value = (const float*)d_in[2];
  // d_in[3] = mask: guaranteed tril -> causality handled analytically
  const float* Wq = (const float*)d_in[4];
  const float* bq = (const float*)d_in[5];
  const float* Wk = (const float*)d_in[6];
  const float* bk = (const float*)d_in[7];
  const float* Wv = (const float*)d_in[8];
  const float* bv = (const float*)d_in[9];
  const float* Wo = (const float*)d_in[10];
  const float* bo = (const float*)d_in[11];

  const size_t WEL = (size_t)D_MODEL * D_MODEL;             // 1Mi
  const size_t ACT = (size_t)SEQ_LEN * BATCH_N * D_MODEL;   // 8Mi

  char* p = (char*)d_ws;
  _Float16* wqh = (_Float16*)p; p += WEL * sizeof(_Float16);
  _Float16* wkh = (_Float16*)p; p += WEL * sizeof(_Float16);
  _Float16* wvh = (_Float16*)p; p += WEL * sizeof(_Float16);
  _Float16* woh = (_Float16*)p; p += WEL * sizeof(_Float16);
  _Float16* qh  = (_Float16*)p; p += ACT * sizeof(_Float16);
  _Float16* kh  = (_Float16*)p; p += ACT * sizeof(_Float16);
  _Float16* vh  = (_Float16*)p; p += ACT * sizeof(_Float16);
  float*    xa  = (float*)p;

  const int M = SEQ_LEN * BATCH_N;

  dim3 cvtBlk(256), cvtGrid((unsigned)(WEL / 4 / 256));
  wcvt_f16<<<cvtGrid, cvtBlk, 0, stream>>>(Wq, wqh, (int)(WEL / 4));
  wcvt_f16<<<cvtGrid, cvtBlk, 0, stream>>>(Wk, wkh, (int)(WEL / 4));
  wcvt_f16<<<cvtGrid, cvtBlk, 0, stream>>>(Wv, wvh, (int)(WEL / 4));
  wcvt_f16<<<cvtGrid, cvtBlk, 0, stream>>>(Wo, woh, (int)(WEL / 4));

  dim3 gBlk(128);
  dim3 gGrid(M / 16, D_MODEL / 256);   // (512, 4)
  gemm_wmma<<<gGrid, gBlk, 0, stream>>>(query, wqh, bq, qh, M, D_MODEL, D_MODEL, 0);
  gemm_wmma<<<gGrid, gBlk, 0, stream>>>(key_,  wkh, bk, kh, M, D_MODEL, D_MODEL, 0);
  gemm_wmma<<<gGrid, gBlk, 0, stream>>>(value, wvh, bv, vh, M, D_MODEL, D_MODEL, 1);

  dim3 aGrid(SEQ_LEN / 64, BATCH_N * N_HEADS);  // (32, 64)
  attn_fwd<<<aGrid, gBlk, 0, stream>>>(qh, kh, vh, xa);

  gemm_wmma<<<gGrid, gBlk, 0, stream>>>(xa, woh, bo, (float*)d_out, M, D_MODEL, D_MODEL, 2);
}
